// GatedRNN_24515673325787
// MI455X (gfx1250) — matrix operand order repaired and checked by reading, not compiled, not run on hardware
//
#include <hip/hip_runtime.h>
#include <hip/hip_bf16.h>
#include <math.h>

typedef _Float16 f16;
typedef __attribute__((ext_vector_type(16))) _Float16 v16h;
typedef __attribute__((ext_vector_type(8)))  _Float16 v8h;
typedef __attribute__((ext_vector_type(8)))  float    v8f;
typedef __attribute__((ext_vector_type(4)))  float    v4f;

#define BB 16384
#define TT 64
#define DD 64
#define HH 256
#define MT 32          // batch rows per workgroup
#define ROWS 264       // padded LDS row stride in halves (528B = 33*16B)
#define XROW 72        // padded x row stride in halves

// packed-weight regions (units: halves). Fragment = 32 lanes * 16 halves = 512.
#define SZ_IH1 (16*2*512)   // 16 ntiles, K=64 -> 2 ktiles
#define SZ_HH  (16*8*512)   // 16 ntiles, K=256 -> 8 ktiles
#define PK_IH1 0
#define PK_HH1 (PK_IH1 + SZ_IH1)
#define PK_IH2 (PK_HH1 + SZ_HH)
#define PK_HH2 (PK_IH2 + SZ_HH)
#define PK_G1  (PK_HH2 + SZ_HH)
#define PK_G2  (PK_G1  + SZ_HH)
#define PK_TOTAL (PK_G2 + SZ_HH)

#if __has_builtin(__builtin_amdgcn_tanhf)
__device__ __forceinline__ float fast_tanh(float x) { return __builtin_amdgcn_tanhf(x); }
#elif __has_builtin(__builtin_amdgcn_tanh_f32)
__device__ __forceinline__ float fast_tanh(float x) { return __builtin_amdgcn_tanh_f32(x); }
#else
__device__ __forceinline__ float fast_tanh(float x) { return tanhf(x); }
#endif

__device__ __forceinline__ float fast_sigmoid(float x) {
    return 1.0f / (1.0f + __expf(-x));
}

union Frag { v16h v; v8h h[2]; };

__device__ __forceinline__ v8f splat8(float v) {
    v8f r;
#pragma unroll
    for (int j = 0; j < 8; ++j) r[j] = v;
    return r;
}

// A-operand fragment (16x32 f16, M x K) from LDS.
// lane l holds row M = (l&15); half i -> K = 16*(i>=8) + (l>>4)*8 + (i&7)
// => two contiguous 8-half (16B) chunks per lane -> ds_load_b128 x2.
__device__ __forceinline__ v16h lds_a(const f16* base, int rowstride, int mt, int kt, int lane) {
    const int r  = mt * 16 + (lane & 15);
    const int k0 = kt * 32 + ((lane >> 4) << 3);
    const f16* p = base + r * rowstride + k0;
    Frag f;
    f.h[0] = *(const v8h*)(p);
    f.h[1] = *(const v8h*)(p + 16);
    return f.v;
}

// B-operand fragment (32x16 f16, K x N) from packed global weights:
// frag = nt*Kt + kt; per lane 16 contiguous halves (32B) -> 2x global_load_b128.
__device__ __forceinline__ v16h ld_b(const f16* P, int regionBase, int Kt, int nt, int kt, int lane) {
    const f16* p = P + regionBase + (((nt * Kt + kt) << 9) + (lane << 4));
    Frag f;
    f.h[0] = *(const v8h*)(p);
    f.h[1] = *(const v8h*)(p + 8);
    return f.v;
}

__device__ __forceinline__ v8f wmma_f16(v16h a, v16h b, v8f c) {
    return __builtin_amdgcn_wmma_f32_16x16x32_f16(
        /*neg_a=*/false, a, /*neg_b=*/false, b,
        /*c_mod=*/(short)0, c, /*reuse_a=*/false, /*reuse_b=*/false);
}

// ---------------------------------------------------------------------------
// Weight pack kernel: f32 -> f16 in WMMA B-fragment layout.
// B[k][n] = W[n][k] (activations multiply W^T). For W_g second half koff=256.
// Fragment element (lane,i): k = kt*32 + 16*(i>>3) + (lane>>4)*8 + (i&7), n = nt*16 + (lane&15).
// ---------------------------------------------------------------------------
__global__ void pack_w(const float* __restrict__ Wih1, const float* __restrict__ Whh1,
                       const float* __restrict__ Wih2, const float* __restrict__ Whh2,
                       const float* __restrict__ Wg,   f16* __restrict__ P) {
    int idx = blockIdx.x * blockDim.x + threadIdx.x;
    if (idx >= PK_TOTAL) return;
    int base, Kt, ld, koff;
    const float* src;
    if (idx < PK_HH1)      { base = PK_IH1; Kt = 2; ld = 64;  koff = 0;   src = Wih1; }
    else if (idx < PK_IH2) { base = PK_HH1; Kt = 8; ld = 256; koff = 0;   src = Whh1; }
    else if (idx < PK_HH2) { base = PK_IH2; Kt = 8; ld = 256; koff = 0;   src = Wih2; }
    else if (idx < PK_G1)  { base = PK_HH2; Kt = 8; ld = 256; koff = 0;   src = Whh2; }
    else if (idx < PK_G2)  { base = PK_G1;  Kt = 8; ld = 512; koff = 0;   src = Wg;   }
    else                   { base = PK_G2;  Kt = 8; ld = 512; koff = 256; src = Wg;   }
    int local  = idx - base;
    int frag   = local >> 9;
    int within = local & 511;
    int lane   = within >> 4;
    int i      = within & 15;
    int kt     = frag % Kt;
    int ntile  = frag / Kt;
    int k = kt * 32 + ((i >> 3) << 4) + ((lane >> 4) << 3) + (i & 7);
    int n = ntile * 16 + (lane & 15);
    P[idx] = (f16)src[n * ld + koff + k];
}

// ---------------------------------------------------------------------------
// Main fused RNN kernel. One workgroup = 32 batch rows, 8 wave32s.
// Wave w owns N-tiles {2w, 2w+1} (columns of H) and both M-tiles.
// ---------------------------------------------------------------------------
__global__ void __launch_bounds__(256)
rnn_main(const float* __restrict__ x,
         const float* __restrict__ b_ih1, const float* __restrict__ b_hh1,
         const float* __restrict__ b_ih2, const float* __restrict__ b_hh2,
         const float* __restrict__ Wo1,   const float* __restrict__ bo1,
         const float* __restrict__ Wo2,   const float* __restrict__ bo2,
         const float* __restrict__ b_g,
         const f16* __restrict__ P, float* __restrict__ out) {
    __shared__ f16 h1s[MT * ROWS];
    __shared__ f16 h2s[MT * ROWS];
    __shared__ f16 us [MT * ROWS];
    __shared__ f16 xs [MT * XROW];
    __shared__ float wo1s[HH];
    __shared__ float wo2s[HH];

    const int tid  = threadIdx.x;
    const int lane = tid & 31;
    const int wid  = tid >> 5;
    const int b0   = blockIdx.x * MT;

    for (int k = tid; k < HH; k += 256) { wo1s[k] = Wo1[k]; wo2s[k] = Wo2[k]; }
    for (int k = tid; k < MT * ROWS; k += 256) {
        h1s[k] = (f16)0.0f; h2s[k] = (f16)0.0f; us[k] = (f16)1.0f;
    }

    // Per-wave column biases (cols = (2w+nn)*16 + lane%16).
    float bias1[2], biasz[2], biasg[2];
#pragma unroll
    for (int nn = 0; nn < 2; ++nn) {
        int col = (2 * wid + nn) * 16 + (lane & 15);
        bias1[nn] = b_ih1[col] + b_hh1[col];
        biasz[nn] = b_ih2[col] + b_hh2[col];
        biasg[nn] = b_g[col];
    }
    const float bo1v = bo1[0];
    const float bo2v = bo2[0];

    const int orow = tid >> 3;   // output pass: 8 lanes per batch row
    const int oseg = tid & 7;

    const int nt0 = 2 * wid;
    const int nt1 = 2 * wid + 1;

    __syncthreads();

    for (int t = 0; t < TT; ++t) {
        // Defeat LICM without destroying pointer provenance: an opaque scalar
        // offset keeps the (L2-hot) weight-fragment loads inside the time loop
        // (no cross-iteration hoist -> no VGPR blowup / scratch spills) while
        // the base still traces to a kernarg, so infer-address-spaces emits
        // global_load_b128 (not flat_load, which would also burn DScnt).
        int tofs = 0;
        asm volatile("" : "+s"(tofs));
        const f16* Pt = P + tofs;

        // --- stage x[b0..b0+31, t, :] -> f16 LDS tile; prefetch next step ---
        {
            int r  = tid >> 3;
            int c0 = (tid & 7) * 8;
            const float* xp = x + ((size_t)(b0 + r) * TT + t) * DD + c0;
            v4f a = *(const v4f*)(xp);
            v4f b = *(const v4f*)(xp + 4);
            v8h hv;
#pragma unroll
            for (int j = 0; j < 4; ++j) { hv[j] = (f16)a[j]; hv[4 + j] = (f16)b[j]; }
            *(v8h*)(xs + r * XROW + c0) = hv;
            if (t + 1 < TT) __builtin_prefetch((const void*)(xp + DD), 0, 0);
        }
        __syncthreads();

        // ---- h1 = tanh(x_t @ Wih1^T + h1 @ Whh1^T + b_ih1 + b_hh1) ----
        v8f acc[2][2];
#pragma unroll
        for (int mt = 0; mt < 2; ++mt)
#pragma unroll
            for (int nn = 0; nn < 2; ++nn) acc[mt][nn] = splat8(bias1[nn]);

#pragma unroll
        for (int kt = 0; kt < 2; ++kt) {         // K over x (64)
            v16h a0 = lds_a(xs, XROW, 0, kt, lane);
            v16h a1 = lds_a(xs, XROW, 1, kt, lane);
            v16h w0 = ld_b(Pt, PK_IH1, 2, nt0, kt, lane);
            v16h w1 = ld_b(Pt, PK_IH1, 2, nt1, kt, lane);
            acc[0][0] = wmma_f16(a0, w0, acc[0][0]);
            acc[0][1] = wmma_f16(a0, w1, acc[0][1]);
            acc[1][0] = wmma_f16(a1, w0, acc[1][0]);
            acc[1][1] = wmma_f16(a1, w1, acc[1][1]);
        }
#pragma unroll
        for (int kt = 0; kt < 8; ++kt) {         // K over h1 (256)
            v16h a0 = lds_a(h1s, ROWS, 0, kt, lane);
            v16h a1 = lds_a(h1s, ROWS, 1, kt, lane);
            v16h w0 = ld_b(Pt, PK_HH1, 8, nt0, kt, lane);
            v16h w1 = ld_b(Pt, PK_HH1, 8, nt1, kt, lane);
            acc[0][0] = wmma_f16(a0, w0, acc[0][0]);
            acc[0][1] = wmma_f16(a0, w1, acc[0][1]);
            acc[1][0] = wmma_f16(a1, w0, acc[1][0]);
            acc[1][1] = wmma_f16(a1, w1, acc[1][1]);
        }
        __syncthreads();   // all reads of old h1s complete

        // tanh + write new h1 (D-tile: lane l, vgpr j -> row mt*16+8*(l>>4)+j, col nt*16+(l&15))
#pragma unroll
        for (int mt = 0; mt < 2; ++mt) {
            int rowb = mt * 16 + ((lane >> 4) << 3);
#pragma unroll
            for (int nn = 0; nn < 2; ++nn) {
                int colb = (2 * wid + nn) * 16 + (lane & 15);
#pragma unroll
                for (int j = 0; j < 8; ++j)
                    h1s[(rowb + j) * ROWS + colb] = (f16)fast_tanh(acc[mt][nn][j]);
            }
        }
        __syncthreads();   // new h1 visible; old h2 still intact

        // ---- z2 = tanh(h1@Wih2^T + h2@Whh2^T + bz); pre-gate g = h1@Wg1^T + h2@Wg2^T + bg ----
        v8f accz[2][2], accu[2][2];
#pragma unroll
        for (int mt = 0; mt < 2; ++mt)
#pragma unroll
            for (int nn = 0; nn < 2; ++nn) { accz[mt][nn] = splat8(biasz[nn]); accu[mt][nn] = splat8(biasg[nn]); }

#pragma unroll 4
        for (int kt = 0; kt < 8; ++kt) {         // pass over new h1
            v16h a0 = lds_a(h1s, ROWS, 0, kt, lane);
            v16h a1 = lds_a(h1s, ROWS, 1, kt, lane);
            v16h z0 = ld_b(Pt, PK_IH2, 8, nt0, kt, lane);
            v16h z1 = ld_b(Pt, PK_IH2, 8, nt1, kt, lane);
            v16h g0 = ld_b(Pt, PK_G1,  8, nt0, kt, lane);
            v16h g1 = ld_b(Pt, PK_G1,  8, nt1, kt, lane);
            accz[0][0] = wmma_f16(a0, z0, accz[0][0]);
            accz[0][1] = wmma_f16(a0, z1, accz[0][1]);
            accz[1][0] = wmma_f16(a1, z0, accz[1][0]);
            accz[1][1] = wmma_f16(a1, z1, accz[1][1]);
            accu[0][0] = wmma_f16(a0, g0, accu[0][0]);
            accu[0][1] = wmma_f16(a0, g1, accu[0][1]);
            accu[1][0] = wmma_f16(a1, g0, accu[1][0]);
            accu[1][1] = wmma_f16(a1, g1, accu[1][1]);
        }
#pragma unroll 4
        for (int kt = 0; kt < 8; ++kt) {         // pass over old h2
            v16h a0 = lds_a(h2s, ROWS, 0, kt, lane);
            v16h a1 = lds_a(h2s, ROWS, 1, kt, lane);
            v16h z0 = ld_b(Pt, PK_HH2, 8, nt0, kt, lane);
            v16h z1 = ld_b(Pt, PK_HH2, 8, nt1, kt, lane);
            v16h g0 = ld_b(Pt, PK_G2,  8, nt0, kt, lane);
            v16h g1 = ld_b(Pt, PK_G2,  8, nt1, kt, lane);
            accz[0][0] = wmma_f16(a0, z0, accz[0][0]);
            accz[0][1] = wmma_f16(a0, z1, accz[0][1]);
            accz[1][0] = wmma_f16(a1, z0, accz[1][0]);
            accz[1][1] = wmma_f16(a1, z1, accz[1][1]);
            accu[0][0] = wmma_f16(a0, g0, accu[0][0]);
            accu[0][1] = wmma_f16(a0, g1, accu[0][1]);
            accu[1][0] = wmma_f16(a1, g0, accu[1][0]);
            accu[1][1] = wmma_f16(a1, g1, accu[1][1]);
        }
        __syncthreads();   // reads of old h2 complete

        // ---- elementwise: h2 = u*z2 + (1-u)*h2 ; u = sigmoid(g) ----
#pragma unroll
        for (int mt = 0; mt < 2; ++mt) {
            int rowb = mt * 16 + ((lane >> 4) << 3);
#pragma unroll
            for (int nn = 0; nn < 2; ++nn) {
                int colb = (2 * wid + nn) * 16 + (lane & 15);
#pragma unroll
                for (int j = 0; j < 8; ++j) {
                    int idx  = (rowb + j) * ROWS + colb;
                    float z  = fast_tanh(accz[mt][nn][j]);
                    float un = fast_sigmoid(accu[mt][nn][j]);
                    float u  = (float)us[idx];
                    float h2o = (float)h2s[idx];
                    float h2n = u * z + (1.0f - u) * h2o;
                    h2s[idx] = (f16)h2n;
                    us[idx]  = (f16)un;
                }
            }
        }
        __syncthreads();   // new h2 visible

        // ---- out[b,t] = tanh(h1.Wo1 + bo1) + tanh(h2.Wo2 + bo2) ----
        {
            float s1 = 0.0f, s2 = 0.0f;
            int kk = oseg * 32;
            const v8h* p1 = (const v8h*)(h1s + orow * ROWS + kk);
            const v8h* p2 = (const v8h*)(h2s + orow * ROWS + kk);
            const v4f* w1 = (const v4f*)(wo1s + kk);
            const v4f* w2 = (const v4f*)(wo2s + kk);
#pragma unroll
            for (int c = 0; c < 4; ++c) {
                v8h hv1 = p1[c];
                v8h hv2 = p2[c];
                v4f wa = w1[2 * c], wb = w1[2 * c + 1];
                v4f wc = w2[2 * c], wd = w2[2 * c + 1];
#pragma unroll
                for (int j = 0; j < 4; ++j) {
                    s1 += (float)hv1[j] * wa[j];
                    s1 += (float)hv1[4 + j] * wb[j];
                    s2 += (float)hv2[j] * wc[j];
                    s2 += (float)hv2[4 + j] * wd[j];
                }
            }
#pragma unroll
            for (int m = 1; m < 8; m <<= 1) {
                s1 += __shfl_xor(s1, m);
                s2 += __shfl_xor(s2, m);
            }
            if (oseg == 0)
                out[(size_t)(b0 + orow) * TT + t] = fast_tanh(s1 + bo1v) + fast_tanh(s2 + bo2v);
        }
        __syncthreads();
    }
}

extern "C" void kernel_launch(void* const* d_in, const int* in_sizes, int n_in,
                              void* d_out, int out_size, void* d_ws, size_t ws_size,
                              hipStream_t stream) {
    (void)in_sizes; (void)n_in; (void)out_size;
    const float* x    = (const float*)d_in[0];
    const float* Wih1 = (const float*)d_in[1];
    const float* bih1 = (const float*)d_in[2];
    const float* Whh1 = (const float*)d_in[3];
    const float* bhh1 = (const float*)d_in[4];
    const float* Wih2 = (const float*)d_in[5];
    const float* bih2 = (const float*)d_in[6];
    const float* Whh2 = (const float*)d_in[7];
    const float* bhh2 = (const float*)d_in[8];
    const float* Wo1  = (const float*)d_in[9];
    const float* bo1  = (const float*)d_in[10];
    const float* Wo2  = (const float*)d_in[11];
    const float* bo2  = (const float*)d_in[12];
    const float* Wg   = (const float*)d_in[13];
    const float* bg   = (const float*)d_in[14];
    f16* P = (f16*)d_ws;
    float* out = (float*)d_out;
    if (ws_size < (size_t)PK_TOTAL * sizeof(f16)) return;

    pack_w<<<(PK_TOTAL + 255) / 256, 256, 0, stream>>>(Wih1, Whh1, Wih2, Whh2, Wg, P);
    rnn_main<<<BB / MT, 256, 0, stream>>>(x, bih1, bhh1, bih2, bhh2,
                                          Wo1, bo1, Wo2, bo2, bg, P, out);
}
